// Conv2d_24833500905755
// MI455X (gfx1250) — matrix operand order repaired and checked by reading, not compiled, not run on hardware
//
#include <hip/hip_runtime.h>

// CDNA5 / gfx1250 conv as 9 shifted GEMMs, double-buffered LDS pipeline.
// M=100352, N=128, K=9x64, bf16 WMMA (f32 accum).

typedef __attribute__((ext_vector_type(16))) __bf16 v16bf;
typedef __attribute__((ext_vector_type(8)))  float  v8f;

#define BATCH 32
#define CIN   64
#define COUT  128
#define HH    56
#define HO    56
#define NSTEP 18               // 9 taps x 2 c-halves

__device__ __forceinline__ unsigned pack_bf16x2(float a, float b) {
    typedef __attribute__((ext_vector_type(2))) float  v2f;
    typedef __attribute__((ext_vector_type(2))) __bf16 v2bf;
    v2f f;
    f.x = a;
    f.y = b;
    union { v2bf h; unsigned u; } p;
    p.h = __builtin_convertvector(f, v2bf);   // single v_cvt_pk_bf16_f32
    return p.u;
}

__global__ __launch_bounds__(256)
void conv_igemm_wmma(const float* __restrict__ x,
                     const float* __restrict__ Kw,   // [576][128] f32 row-major
                     float* __restrict__ out) {
    // Double-buffered bf16 tiles (dword-packed): A/B = 128 x 32, 8KB each.
    __shared__ __attribute__((aligned(16))) unsigned Ash[2][128 * 16];
    __shared__ __attribute__((aligned(16))) unsigned Bsh[2][128 * 16];

    const int tid = threadIdx.x;
    const int m0  = blockIdx.x * 128;

    const int lane  = tid & 31;
    const int w     = tid >> 5;
    const int lrow  = lane & 15;
    const int lhalf = lane >> 4;
    const int mw    = (w & 1) * 64;
    const int nw    = (w >> 1) * 32;

    v8f acc[4][2] = {};

    // A staging: 2 threads per row, 16 consecutive channels each
    const int arow = tid & 127;
    const int ako  = (tid >> 7) * 16;
    const int am   = m0 + arow;
    const int ab   = am / (HO * HO);
    const int arem = am % (HO * HO);
    const int ai   = arem / HO;
    const int aj   = arem % HO;
    const float* xb = x + (size_t)ab * CIN * HH * HH + (size_t)ako * (HH * HH);

    // B staging: 2 threads per output channel, 16 consecutive channels each
    const int bn  = tid & 127;
    const int bko = (tid >> 7) * 16;

    float av[16], bv[16];

    // ---- staging: global -> regs for step s ----
    auto LOAD = [&](int s) {
        const int r9 = s >> 1;                 // tap (scalar math)
        const int c0 = (s & 1) * 32;
        const int di = r9 / 3;
        const int dj = r9 - di * 3;
        const int ii = ai + di - 1;
        const int jj = aj + dj - 1;
        const bool inb = ((unsigned)ii < (unsigned)HH) && ((unsigned)jj < (unsigned)HH);
        const float* asrc = xb + (size_t)c0 * (HH * HH) + ii * HH + jj;
        if (inb) {
            #pragma unroll
            for (int t = 0; t < 16; ++t) av[t] = asrc[t * (HH * HH)];
        } else {
            #pragma unroll
            for (int t = 0; t < 16; ++t) av[t] = 0.0f;
        }
        const float* bsrc = Kw + (size_t)(c0 + bko) * 9 * COUT + r9 * COUT + bn;
        #pragma unroll
        for (int t = 0; t < 16; ++t) bv[t] = bsrc[t * 9 * COUT];
    };

    // ---- staging: regs -> LDS buffer p (pack f32 pairs to bf16) ----
    auto STORE = [&](int p) {
        #pragma unroll
        for (int t = 0; t < 8; ++t)
            Ash[p][arow * 16 + (ako >> 1) + t] = pack_bf16x2(av[2 * t], av[2 * t + 1]);
        #pragma unroll
        for (int t = 0; t < 8; ++t)
            Bsh[p][bn * 16 + (bko >> 1) + t] = pack_bf16x2(bv[2 * t], bv[2 * t + 1]);
    };

    // ---- compute: fragments from buffer p + 8 WMMAs ----
    auto COMPUTE = [&](int p) {
        union Frag { unsigned u[8]; v16bf v; };
        Frag afr[4], bfr[2];
        #pragma unroll
        for (int s = 0; s < 4; ++s) {
            const unsigned* ap = &Ash[p][(mw + s * 16 + lrow) * 16];
            #pragma unroll
            for (int vv = 0; vv < 8; ++vv) {
                const int dw = (vv < 4) ? (lhalf * 4 + vv) : (8 + lhalf * 4 + (vv - 4));
                afr[s].u[vv] = ap[dw];
            }
        }
        #pragma unroll
        for (int t = 0; t < 2; ++t) {
            const unsigned* bp = &Bsh[p][(nw + t * 16 + lrow) * 16];
            #pragma unroll
            for (int vv = 0; vv < 8; ++vv)
                bfr[t].u[vv] = bp[(vv < 4) ? (lhalf * 4 + vv) : (8 + lhalf * 4 + (vv - 4))];
        }
        #pragma unroll
        for (int s = 0; s < 4; ++s)
            #pragma unroll
            for (int t = 0; t < 2; ++t)
                acc[s][t] = __builtin_amdgcn_wmma_f32_16x16x32_bf16(
                    false, afr[s].v, false, bfr[t].v,
                    (short)0, acc[s][t], false, false);
    };

    // ---- pipelined main loop: one barrier per step ----
    LOAD(0);
    STORE(0);
    __syncthreads();
    for (int s = 0; s < NSTEP; ++s) {
        if (s + 1 < NSTEP) LOAD(s + 1);      // global loads overlap WMMAs below
        COMPUTE(s & 1);
        if (s + 1 < NSTEP) STORE((s + 1) & 1);
        __syncthreads();
    }

    // ---- epilogue: one base addr per (s,t); cheap boundary fix per row ----
    #pragma unroll
    for (int s = 0; s < 4; ++s) {
        const int mmb = m0 + mw + s * 16 + lhalf * 8;        // 8 consecutive m
        const int b   = mmb / (HO * HO);
        const int rm  = mmb % (HO * HO);
        #pragma unroll
        for (int t = 0; t < 2; ++t) {
            const int n    = nw + t * 16 + lrow;
            const int base = mmb + (HO * HO) * (n + b * (COUT - 1));
            #pragma unroll
            for (int r = 0; r < 8; ++r) {
                const int fix = ((rm + r) >= (HO * HO)) ? ((COUT - 1) * HO * HO) : 0;
                out[base + r + fix] = acc[s][t][r];
            }
        }
    }
}

extern "C" void kernel_launch(void* const* d_in, const int* in_sizes, int n_in,
                              void* d_out, int out_size, void* d_ws, size_t ws_size,
                              hipStream_t stream) {
    (void)in_sizes; (void)n_in; (void)d_ws; (void)ws_size; (void)out_size;
    const float* x  = (const float*)d_in[0];   // [32,64,56,56] f32
    const float* Kw = (const float*)d_in[1];   // [64,3,3,128] f32 == [576][128]
    float* out      = (float*)d_out;           // [32,128,56,56] f32

    const int mtiles = (BATCH * HO * HO) / 128;   // 784, exact
    conv_igemm_wmma<<<mtiles, 256, 0, stream>>>(x, Kw, out);
}
